// MambaObserverClassifier_39565238731518
// MI455X (gfx1250) — matrix operand order repaired and checked by reading, not compiled, not run on hardware
//
#include <hip/hip_runtime.h>

typedef __attribute__((ext_vector_type(16))) __bf16 v16bf;
typedef __attribute__((ext_vector_type(8)))  float  v8f;

#define B_SZ   128
#define IMG    28
#define GRID_  7
#define L_SEQ  49
#define DM     512
#define NL     8
#define DS_    16
#define DI     1024
#define NC     10
#define NTOK   (B_SZ * L_SEQ)   // 6272

// CDNA5 async global->LDS copy path (ASYNCcnt), guarded so the file compiles
// on toolchains that haven't exposed the builtins yet.
#if defined(__gfx1250__) && \
    __has_builtin(__builtin_amdgcn_global_load_async_to_lds_b128) && \
    __has_builtin(__builtin_amdgcn_s_wait_asynccnt)
#define USE_ASYNC_LDS 1
#else
#define USE_ASYNC_LDS 0
#endif

#if USE_ASYNC_LDS
// Probe-derived signature: param0 = AS1 pointer to int __vector(4) (non-const),
// param1 = LDS pointer, then imm offset + imm cpol.
typedef int v4i_a __attribute__((vector_size(16)));
typedef __attribute__((address_space(1))) v4i_a* async_gptr_t;
typedef __attribute__((address_space(3))) v4i_a* async_lptr_t;

__device__ __forceinline__ void async_copy_b128(const void* g, void* l) {
  __builtin_amdgcn_global_load_async_to_lds_b128(
      (async_gptr_t)(g), (async_lptr_t)(l), 0, 0);
}
#endif

// ---------------------------------------------------------------------------
// fp32 -> bf16 weight staging, TRANSPOSED to K-major: dst[n*K + k] = src[k*N + n]
// (B operand of every GEMM is a weight; K-major makes both LDS tiles
//  contiguous 16B chunks -> b128 async copies, no transpose-scatter.)
// grid.y = layer index
// ---------------------------------------------------------------------------
__global__ void f32_to_bf16_T_kernel(const float* __restrict__ src,
                                     __bf16* __restrict__ dst, int K, int N) {
  size_t layerOff = (size_t)blockIdx.y * K * N;
  int i = blockIdx.x * blockDim.x + threadIdx.x;   // n*K + k within layer
  if (i >= K * N) return;
  int k = i % K, n = i / K;
  dst[layerOff + i] = (__bf16)src[layerOff + (size_t)k * N + n];
}

// ---------------------------------------------------------------------------
// Patch embedding: h[b,l,d] = sum_pp patch * pw[d,pp] + pb[d] + pos[l,d]
// ---------------------------------------------------------------------------
__global__ void patch_embed_kernel(const float* __restrict__ x,
                                   const float* __restrict__ pw,
                                   const float* __restrict__ pb,
                                   const float* __restrict__ pos,
                                   float* __restrict__ h) {
  int idx = blockIdx.x * blockDim.x + threadIdx.x;   // (b*L + l)*DM + d
  if (idx >= NTOK * DM) return;
  int d   = idx & (DM - 1);
  int tok = idx / DM;
  int l   = tok % L_SEQ;
  int b   = tok / L_SEQ;
  int gy = l / GRID_, gx = l % GRID_;
  const float* img = x + (size_t)b * IMG * IMG;
  const float* w   = pw + (size_t)d * 16;
  float acc = pb[d];
#pragma unroll
  for (int py = 0; py < 4; ++py)
#pragma unroll
    for (int px = 0; px < 4; ++px)
      acc += img[(gy * 4 + py) * IMG + gx * 4 + px] * w[py * 4 + px];
  h[idx] = acc + pos[(size_t)l * DM + d];
}

// ---------------------------------------------------------------------------
// LayerNorm over DM=512, one wave32 per token, bf16 output for WMMA A operand
// ---------------------------------------------------------------------------
__global__ __launch_bounds__(256)
void layernorm_bf16_kernel(const float* __restrict__ x,
                           const float* __restrict__ g,
                           const float* __restrict__ b,
                           __bf16* __restrict__ out, int ntok) {
  int wave = threadIdx.x >> 5;
  int lane = threadIdx.x & 31;
  int tok  = blockIdx.x * 8 + wave;
  if (tok >= ntok) return;
  const float* xr = x + (size_t)tok * DM;
  float vals[16];
  float s = 0.f, ss = 0.f;
#pragma unroll
  for (int j = 0; j < 16; ++j) {
    float v = xr[lane + j * 32];
    vals[j] = v; s += v; ss += v * v;
  }
#pragma unroll
  for (int off = 16; off >= 1; off >>= 1) {
    s  += __shfl_xor(s, off);
    ss += __shfl_xor(ss, off);
  }
  float m   = s * (1.f / DM);
  float var = ss * (1.f / DM) - m * m;
  float inv = rsqrtf(var + 1e-5f);
  __bf16* orow = out + (size_t)tok * DM;
#pragma unroll
  for (int j = 0; j < 16; ++j) {
    int c = lane + j * 32;
    orow[c] = (__bf16)((vals[j] - m) * inv * g[c] + b[c]);
  }
}

// ---------------------------------------------------------------------------
// bf16 WMMA GEMM: C[M,N] (fp32) = A[M,K] @ Bt[N,K]^T   (flags bit0: C += r)
// Block tile 128x64, K-step 32, 8 waves; wave = 16 rows x 4 N-tiles
// -> 4x v_wmma_f32_16x16x32_bf16 per K-step per wave.
// Double-buffered LDS tiles; async global->LDS copies when available.
// LDS stride 40 elems (80 B): 16B-aligned rows for b128, conflict-free
// fragment reads (20-dword row stride spans all 64 banks).
// Requires: M % 128 == 0, N % 64 == 0, K % 32 == 0.
// ---------------------------------------------------------------------------
__global__ __launch_bounds__(256)
void gemm_bf16_wmma_kernel(const __bf16* __restrict__ A,
                           const __bf16* __restrict__ Bt,
                           float* __restrict__ C,
                           int M, int N, int K, int flags) {
  __shared__ __bf16 As[2][128][40];
  __shared__ __bf16 Bs[2][64][40];

  int tid  = threadIdx.x;
  int lane = tid & 31;
  int wave = tid >> 5;
  int mBase = blockIdx.y * 128;
  int nBase = blockIdx.x * 64;

  // per-thread tile-staging coordinates (contiguous 16B chunks)
  int arow = tid >> 1, acol = (tid & 1) * 16;   // 128 rows x 32 k
  int brow = tid >> 2, bcol = (tid & 3) * 8;    // 64 rows  x 32 k

  const __bf16* gA = A  + (size_t)(mBase + arow) * K + acol;
  const __bf16* gB = Bt + (size_t)(nBase + brow) * K + bcol;

  v8f acc0 = {}, acc1 = {}, acc2 = {}, acc3 = {};

  int khalfA = (lane < 16) ? 0 : 8;    // ISA 7.12.2 A 16-bit 16x32 layout
  int khalfB = (lane < 16) ? 0 : 8;    // ISA 7.12.2 B 16-bit 32x16 layout
  int rA = wave * 16 + (lane & 15);
  int cB = lane & 15;

  // ---- prologue: stage tile 0 into buffer 0 ----
#if USE_ASYNC_LDS
  async_copy_b128(gA,      &As[0][arow][acol]);
  async_copy_b128(gA + 8,  &As[0][arow][acol + 8]);
  async_copy_b128(gB,      &Bs[0][brow][bcol]);
  __builtin_amdgcn_s_wait_asynccnt(0);
#else
  {
    uint4 a0 = *(const uint4*)(gA);
    uint4 a1 = *(const uint4*)(gA + 8);
    uint4 b0 = *(const uint4*)(gB);
    *(uint4*)&As[0][arow][acol]     = a0;
    *(uint4*)&As[0][arow][acol + 8] = a1;
    *(uint4*)&Bs[0][brow][bcol]     = b0;
  }
#endif
  __syncthreads();

  for (int k0 = 0; k0 < K; k0 += 32) {
    int buf = (k0 >> 5) & 1;
    int nxt = buf ^ 1;
    bool more = (k0 + 32) < K;

    // ---- start staging tile k0+32 into the other buffer ----
#if USE_ASYNC_LDS
    if (more) {
      async_copy_b128(gA + k0 + 32, &As[nxt][arow][acol]);
      async_copy_b128(gA + k0 + 40, &As[nxt][arow][acol + 8]);
      async_copy_b128(gB + k0 + 32, &Bs[nxt][brow][bcol]);
    }
#else
    uint4 na0{}, na1{}, nb0{};
    if (more) {
      na0 = *(const uint4*)(gA + k0 + 32);
      na1 = *(const uint4*)(gA + k0 + 40);
      nb0 = *(const uint4*)(gB + k0 + 32);
    }
#endif
    if (k0 + 64 < K) {   // gfx1250 global_prefetch of the tile after next
      __builtin_prefetch(gA + k0 + 64, 0, 1);
      __builtin_prefetch(gB + k0 + 64, 0, 1);
    }

    // ---- A fragment (wave's 16x32 sub-tile) ----
    union { v16bf v; unsigned u[8]; } af;
#pragma unroll
    for (int j = 0; j < 8; ++j) {
      int k = ((j >> 2) << 4) + khalfA + ((j & 3) << 1);
      af.u[j] = *(const unsigned*)&As[buf][rA][k];
    }
    // ---- 4 B fragments + WMMA ----
    union { v16bf v; unsigned u[8]; } b0, b1, b2, b3;
#pragma unroll
    for (int j = 0; j < 8; ++j) {
      int k = ((j >> 2) << 4) + khalfB + ((j & 3) << 1);
      b0.u[j] = *(const unsigned*)&Bs[buf][ 0 + cB][k];
      b1.u[j] = *(const unsigned*)&Bs[buf][16 + cB][k];
      b2.u[j] = *(const unsigned*)&Bs[buf][32 + cB][k];
      b3.u[j] = *(const unsigned*)&Bs[buf][48 + cB][k];
    }
    acc0 = __builtin_amdgcn_wmma_f32_16x16x32_bf16(false, af.v, false, b0.v, (short)0, acc0, false, false);
    acc1 = __builtin_amdgcn_wmma_f32_16x16x32_bf16(false, af.v, false, b1.v, (short)0, acc1, false, false);
    acc2 = __builtin_amdgcn_wmma_f32_16x16x32_bf16(false, af.v, false, b2.v, (short)0, acc2, false, false);
    acc3 = __builtin_amdgcn_wmma_f32_16x16x32_bf16(false, af.v, false, b3.v, (short)0, acc3, false, false);

    // ---- finish staging next tile, rotate buffers ----
#if USE_ASYNC_LDS
    __builtin_amdgcn_s_wait_asynccnt(0);
#else
    if (more) {
      *(uint4*)&As[nxt][arow][acol]     = na0;
      *(uint4*)&As[nxt][arow][acol + 8] = na1;
      *(uint4*)&Bs[nxt][brow][bcol]     = nb0;
    }
#endif
    __syncthreads();
  }

  // ---- store C per ISA C/D layout: vgpr v -> M = v + (lane<16 ? 0 : 8) ----
  int rowTop = mBase + wave * 16 + ((lane < 16) ? 0 : 8);
  int colIn  = lane & 15;
  v8f accs[4] = {acc0, acc1, acc2, acc3};
#pragma unroll
  for (int nt = 0; nt < 4; ++nt) {
    int col = nBase + nt * 16 + colIn;
#pragma unroll
    for (int v = 0; v < 8; ++v) {
      size_t idx = (size_t)(rowTop + v) * N + col;
      float val = accs[nt][v];
      if (flags & 1) C[idx] += val; else C[idx] = val;
    }
  }
}

// ---------------------------------------------------------------------------
// Causal conv1d (D_CONV=4) + SiLU over the xp half of xz; emits fp32 + bf16
// ---------------------------------------------------------------------------
__global__ __launch_bounds__(256)
void conv_silu_kernel(const float* __restrict__ xz,
                      const float* __restrict__ cw,
                      const float* __restrict__ cb,
                      float* __restrict__ xp_f,
                      __bf16* __restrict__ xp_b) {
  int idx = blockIdx.x * blockDim.x + threadIdx.x;   // b*DI + d
  if (idx >= B_SZ * DI) return;
  int d = idx & (DI - 1);
  int b = idx / DI;
  float w0 = cw[d * 4 + 0], w1 = cw[d * 4 + 1], w2 = cw[d * 4 + 2], w3 = cw[d * 4 + 3];
  float bias = cb[d];
  float xm3 = 0.f, xm2 = 0.f, xm1 = 0.f;
  for (int t = 0; t < L_SEQ; ++t) {
    size_t tok = (size_t)(b * L_SEQ + t);
    float x0 = xz[tok * (2 * DI) + d];
    float y  = bias + w0 * xm3 + w1 * xm2 + w2 * xm1 + w3 * x0;
    y = y / (1.f + __expf(-y));          // SiLU
    xp_f[tok * DI + d] = y;
    xp_b[tok * DI + d] = (__bf16)y;
    xm3 = xm2; xm2 = xm1; xm1 = x0;
  }
}

// ---------------------------------------------------------------------------
// dt = softplus(dt_lo @ Wdt + bdt); K=32 so plain VALU dot is fine
// ---------------------------------------------------------------------------
__global__ void dt_softplus_kernel(const float* __restrict__ dbc,
                                   const float* __restrict__ Wdt,
                                   const float* __restrict__ bdt,
                                   float* __restrict__ dt) {
  int idx = blockIdx.x * blockDim.x + threadIdx.x;   // tok*DI + d
  if (idx >= NTOK * DI) return;
  int d   = idx & (DI - 1);
  int tok = idx / DI;
  float acc = bdt[d];
  const float* db = dbc + (size_t)tok * 64;
#pragma unroll 8
  for (int r = 0; r < 32; ++r) acc += db[r] * Wdt[(size_t)r * DI + d];
  dt[idx] = (acc > 20.f) ? acc : log1pf(__expf(acc));
}

// ---------------------------------------------------------------------------
// Selective scan + SiLU(z) gating; one thread per (b, d), 16-float state in
// registers, per-step B/C vectors staged in LDS. grid = (B_SZ, DI/256)
// ---------------------------------------------------------------------------
__global__ __launch_bounds__(256)
void selective_scan_kernel(const float* __restrict__ xp_f,
                           const float* __restrict__ dt,
                           const float* __restrict__ dbc,
                           const float* __restrict__ xz,
                           const float* __restrict__ A_log,
                           const float* __restrict__ Dp,
                           __bf16* __restrict__ y_b) {
  __shared__ float BC[32];                 // [0..15]=B_t, [16..31]=C_t
  int b = blockIdx.x;
  int d = blockIdx.y * 256 + threadIdx.x;
  float A[DS_];
#pragma unroll
  for (int n = 0; n < DS_; ++n) A[n] = -__expf(A_log[(size_t)d * DS_ + n]);
  float Dd = Dp[d];
  float h[DS_] = {};
  for (int t = 0; t < L_SEQ; ++t) {
    size_t tok = (size_t)b * L_SEQ + t;
    __syncthreads();
    if (threadIdx.x < 32) BC[threadIdx.x] = dbc[tok * 64 + 32 + threadIdx.x];
    __syncthreads();
    float dtv = dt[tok * DI + d];
    float xv  = xp_f[tok * DI + d];
    float y = 0.f;
#pragma unroll
    for (int n = 0; n < DS_; ++n) {
      float dA = __expf(dtv * A[n]);
      h[n] = dA * h[n] + dtv * BC[n] * xv;
      y += h[n] * BC[16 + n];
    }
    y += Dd * xv;
    float z = xz[tok * (2 * DI) + DI + d];
    y *= z / (1.f + __expf(-z));           // gate with SiLU(z)
    y_b[tok * DI + d] = (__bf16)y;
  }
}

// ---------------------------------------------------------------------------
// Mean-pool over L then LayerNorm; one block per batch element
// ---------------------------------------------------------------------------
__global__ __launch_bounds__(256)
void pool_norm_kernel(const float* __restrict__ h,
                      const float* __restrict__ g,
                      const float* __restrict__ bb,
                      float* __restrict__ out) {
  __shared__ float rs[8], rss[8], mv[2];
  int b = blockIdx.x;
  int tid = threadIdx.x;
  float p0, p1;
  {
    float s0 = 0.f, s1 = 0.f;
    for (int t = 0; t < L_SEQ; ++t) {
      const float* row = h + ((size_t)b * L_SEQ + t) * DM;
      s0 += row[tid];
      s1 += row[tid + 256];
    }
    p0 = s0 * (1.f / L_SEQ);
    p1 = s1 * (1.f / L_SEQ);
  }
  float s  = p0 + p1;
  float ss = p0 * p0 + p1 * p1;
  int lane = tid & 31, wave = tid >> 5;
#pragma unroll
  for (int off = 16; off >= 1; off >>= 1) {
    s  += __shfl_xor(s, off);
    ss += __shfl_xor(ss, off);
  }
  if (lane == 0) { rs[wave] = s; rss[wave] = ss; }
  __syncthreads();
  if (tid == 0) {
    float S = 0.f, SS = 0.f;
    for (int w = 0; w < 8; ++w) { S += rs[w]; SS += rss[w]; }
    float m = S * (1.f / DM);
    mv[0] = m;
    mv[1] = rsqrtf(SS * (1.f / DM) - m * m + 1e-5f);
  }
  __syncthreads();
  float m = mv[0], inv = mv[1];
  out[(size_t)b * DM + tid]       = (p0 - m) * inv * g[tid]       + bb[tid];
  out[(size_t)b * DM + tid + 256] = (p1 - m) * inv * g[tid + 256] + bb[tid + 256];
}

// ---------------------------------------------------------------------------
// Head: hid = gelu(pooled @ W1 + b1) [128,256]; out = hid @ W2 + b2 [128,10]
// ---------------------------------------------------------------------------
__global__ void head1_kernel(const float* __restrict__ pooled,
                             const float* __restrict__ W1,
                             const float* __restrict__ b1,
                             float* __restrict__ hid) {
  int idx = blockIdx.x * blockDim.x + threadIdx.x;   // b*256 + j
  if (idx >= B_SZ * 256) return;
  int j = idx & 255;
  int b = idx >> 8;
  const float* pr = pooled + (size_t)b * DM;
  float acc = b1[j];
  for (int k = 0; k < DM; ++k) acc += pr[k] * W1[(size_t)k * 256 + j];
  hid[idx] = 0.5f * acc * (1.f + erff(acc * 0.7071067811865475f));  // exact gelu
}

__global__ void head2_kernel(const float* __restrict__ hid,
                             const float* __restrict__ W2,
                             const float* __restrict__ b2,
                             float* __restrict__ out) {
  int idx = blockIdx.x * blockDim.x + threadIdx.x;   // b*NC + c
  if (idx >= B_SZ * NC) return;
  int c = idx % NC;
  int b = idx / NC;
  const float* hr = hid + (size_t)b * 256;
  float acc = b2[c];
  for (int k = 0; k < 256; ++k) acc += hr[k] * W2[k * NC + c];
  out[idx] = acc;
}

// ---------------------------------------------------------------------------
// Host orchestration
// ---------------------------------------------------------------------------
extern "C" void kernel_launch(void* const* d_in, const int* in_sizes, int n_in,
                              void* d_out, int out_size, void* d_ws, size_t ws_size,
                              hipStream_t stream) {
  (void)in_sizes; (void)n_in; (void)out_size; (void)ws_size;
  const float* x       = (const float*)d_in[0];
  const float* patch_w = (const float*)d_in[1];
  const float* patch_b = (const float*)d_in[2];
  const float* pos     = (const float*)d_in[3];
  const float* Win     = (const float*)d_in[4];
  const float* conv_w  = (const float*)d_in[5];
  const float* conv_b  = (const float*)d_in[6];
  const float* Wx      = (const float*)d_in[7];
  const float* Wdt     = (const float*)d_in[8];
  const float* bdt     = (const float*)d_in[9];
  const float* A_log   = (const float*)d_in[10];
  const float* Dp      = (const float*)d_in[11];
  const float* Wout    = (const float*)d_in[12];
  const float* ln_g    = (const float*)d_in[13];
  const float* ln_b    = (const float*)d_in[14];
  const float* norm_g  = (const float*)d_in[15];
  const float* norm_b  = (const float*)d_in[16];
  const float* W1      = (const float*)d_in[17];
  const float* b1      = (const float*)d_in[18];
  const float* W2      = (const float*)d_in[19];
  const float* b2      = (const float*)d_in[20];
  float* out = (float*)d_out;

  char* ws = (char*)d_ws;
  size_t off = 0;
  auto alloc = [&](size_t bytes) -> char* {
    char* p = ws + off;
    off = (off + bytes + 255) & ~(size_t)255;
    return p;
  };
  float*  h      = (float*) alloc((size_t)NTOK * DM * 4);
  __bf16* hn     = (__bf16*)alloc((size_t)NTOK * DM * 2);
  float*  xz     = (float*) alloc((size_t)NTOK * 2 * DI * 4);
  float*  xp_f   = (float*) alloc((size_t)NTOK * DI * 4);
  __bf16* xp_b   = (__bf16*)alloc((size_t)NTOK * DI * 2);
  float*  dbc    = (float*) alloc((size_t)NTOK * 64 * 4);
  float*  dtb    = (float*) alloc((size_t)NTOK * DI * 4);
  __bf16* y_b    = (__bf16*)alloc((size_t)NTOK * DI * 2);
  float*  pooled = (float*) alloc((size_t)B_SZ * DM * 4);
  float*  hid    = (float*) alloc((size_t)B_SZ * 256 * 4);
  __bf16* Win_t  = (__bf16*)alloc((size_t)NL * DM * 2 * DI * 2);   // [2DI][DM] per layer
  __bf16* Wx_t   = (__bf16*)alloc((size_t)NL * DI * 64 * 2);       // [64][DI]
  __bf16* Wout_t = (__bf16*)alloc((size_t)NL * DI * DM * 2);       // [DM][DI]

  // Stage all GEMM weights as bf16, K-major (transposed), once per call.
  f32_to_bf16_T_kernel<<<dim3((DM * 2 * DI) / 256, NL), 256, 0, stream>>>(
      Win, Win_t, DM, 2 * DI);
  f32_to_bf16_T_kernel<<<dim3((DI * 64) / 256, NL), 256, 0, stream>>>(
      Wx, Wx_t, DI, 64);
  f32_to_bf16_T_kernel<<<dim3((DI * DM) / 256, NL), 256, 0, stream>>>(
      Wout, Wout_t, DI, DM);

  patch_embed_kernel<<<(NTOK * DM) / 256, 256, 0, stream>>>(x, patch_w, patch_b, pos, h);

  for (int i = 0; i < NL; ++i) {
    const __bf16* Win_i  = Win_t  + (size_t)i * DM * 2 * DI;
    const __bf16* Wx_i   = Wx_t   + (size_t)i * DI * 64;
    const __bf16* Wout_i = Wout_t + (size_t)i * DI * DM;

    // 1) LayerNorm -> bf16
    layernorm_bf16_kernel<<<NTOK / 8, 256, 0, stream>>>(
        h, ln_g + (size_t)i * DM, ln_b + (size_t)i * DM, hn, NTOK);

    // 2) xz = hn @ Win  : [6272,512]x[512,2048]
    gemm_bf16_wmma_kernel<<<dim3(2 * DI / 64, NTOK / 128), 256, 0, stream>>>(
        hn, Win_i, xz, NTOK, 2 * DI, DM, 0);

    // 3) causal conv1d + SiLU
    conv_silu_kernel<<<(B_SZ * DI) / 256, 256, 0, stream>>>(
        xz, conv_w + (size_t)i * DI * 4, conv_b + (size_t)i * DI, xp_f, xp_b);

    // 4) dbc = xp @ Wx : [6272,1024]x[1024,64]
    gemm_bf16_wmma_kernel<<<dim3(1, NTOK / 128), 256, 0, stream>>>(
        xp_b, Wx_i, dbc, NTOK, 64, DI, 0);

    // 5) dt = softplus(dt_lo @ Wdt + bdt)
    dt_softplus_kernel<<<(NTOK * DI) / 256, 256, 0, stream>>>(
        dbc, Wdt + (size_t)i * 32 * DI, bdt + (size_t)i * DI, dtb);

    // 6) selective scan + gating -> y (bf16)
    selective_scan_kernel<<<dim3(B_SZ, DI / 256), 256, 0, stream>>>(
        xp_f, dtb, dbc, xz, A_log + (size_t)i * DI * DS_, Dp + (size_t)i * DI, y_b);

    // 7) h += y @ Wout : [6272,1024]x[1024,512], residual accumulate
    gemm_bf16_wmma_kernel<<<dim3(DM / 64, NTOK / 128), 256, 0, stream>>>(
        y_b, Wout_i, h, NTOK, DM, DI, 1);
  }

  pool_norm_kernel<<<B_SZ, 256, 0, stream>>>(h, norm_g, norm_b, pooled);
  head1_kernel<<<(B_SZ * 256) / 256, 256, 0, stream>>>(pooled, W1, b1, hid);
  head2_kernel<<<(B_SZ * NC + 255) / 256, 256, 0, stream>>>(hid, W2, b2, out);
}
